// AttentionPooling_26414048870849
// MI455X (gfx1250) — compile-verified, matrix operand order checked by reading, not run on hardware
//
#include <hip/hip_runtime.h>
#include <hip/hip_bf16.h>
#include <math.h>

#define B_  16
#define T_  8192
#define H_  256
#define K_  128
#define HQ_ 64

typedef __attribute__((ext_vector_type(16))) __bf16 v16bf;
typedef __attribute__((ext_vector_type(8)))  __bf16 v8bf;
typedef __attribute__((ext_vector_type(8)))  float  v8f;
typedef __attribute__((ext_vector_type(4)))  float  v4f;

// ---------------------------------------------------------------------------
// Kernel 0: W1 (H x HQ, f32, row-major) -> W1T (HQ x H, bf16, row-major)
// so the WMMA B-operand per lane is two contiguous 16B loads.
// ---------------------------------------------------------------------------
__global__ void __launch_bounds__(256)
w1_transpose_kernel(const float* __restrict__ W1, __bf16* __restrict__ W1T) {
    const int n = blockIdx.x;      // 0..HQ-1
    const int k = threadIdx.x;     // 0..H-1
    W1T[n * H_ + k] = (__bf16)W1[k * HQ_ + n];
}

// ---------------------------------------------------------------------------
// Kernel 1: scores[b,t] = tanh(X@W1+b1) @ W2 + b2 via bf16 WMMA (f32 accum).
// One wave per 16-token tile; 8 waves (256 thr) per block.
// A-operand layout (16-bit A 16x32, per ISA 7.12.2):
//   lane<16 : M=lane,  elements = K{k0+0..7, k0+16..23}
//   lane>=16: M=lane-16, elements = K{k0+8..15, k0+24..31}
// C/D layout: VGPR v holds M = v + 8*(lane>=16), N = lane&15.
// ---------------------------------------------------------------------------
__global__ void __launch_bounds__(256)
scores_wmma_kernel(const float*  __restrict__ X,
                   const __bf16* __restrict__ W1T,
                   const float*  __restrict__ b1,
                   const float*  __restrict__ W2,
                   const float*  __restrict__ b2,
                   float*        __restrict__ scores) {
    const int lane = threadIdx.x & 31;
    const int wv   = threadIdx.x >> 5;
    const int half = lane >> 4;
    const int l16  = lane & 15;
    const int mrow = (blockIdx.x * 8 + wv) * 16;          // token-tile base in [0, B*T)

    v8f acc[4] = {};                                      // 4 n-tiles of 16 -> HQ=64

    const float* Xrow = X + (size_t)(mrow + l16) * H_;

    #pragma unroll
    for (int step = 0; step < H_ / 32; ++step) {
        const int k0 = step * 32;
        const int g0 = k0 + half * 8;
        const int g1 = g0 + 16;

        // ---- A operand: 16 f32 -> bf16 (two contiguous 32B groups) ----
        const v4f* p0 = (const v4f*)(Xrow + g0);
        const v4f* p1 = (const v4f*)(Xrow + g1);
        v4f f0 = p0[0], f1 = p0[1], f2 = p1[0], f3 = p1[1];
        v16bf a;
        #pragma unroll
        for (int e = 0; e < 4; ++e) {
            a[e]      = (__bf16)f0[e];
            a[4 + e]  = (__bf16)f1[e];
            a[8 + e]  = (__bf16)f2[e];
            a[12 + e] = (__bf16)f3[e];
        }

        // ---- B operands: W1T rows are contiguous in K ----
        #pragma unroll
        for (int nt = 0; nt < 4; ++nt) {
            const int n = nt * 16 + l16;
            const __bf16* wp = W1T + n * H_;
            v8bf blo = *(const v8bf*)(wp + g0);
            v8bf bhi = *(const v8bf*)(wp + g1);
            v16bf bb = __builtin_shufflevector(blo, bhi,
                        0,1,2,3,4,5,6,7,8,9,10,11,12,13,14,15);
            acc[nt] = __builtin_amdgcn_wmma_f32_16x16x32_bf16(
                          false, a, false, bb, (short)0, acc[nt], false, false);
        }
    }

    // ---- epilogue: tanh, dot with W2 (64 wide), reduce across 16 lanes ----
    float p[8];
    #pragma unroll
    for (int v = 0; v < 8; ++v) p[v] = 0.f;

    #pragma unroll
    for (int nt = 0; nt < 4; ++nt) {
        const int n   = nt * 16 + l16;
        const float b1n = b1[n];
        const float w2n = W2[n];
        #pragma unroll
        for (int v = 0; v < 8; ++v)
            p[v] += tanhf(acc[nt][v] + b1n) * w2n;
    }

    // butterfly over lanes 0..15 within each 16-lane half (bit4 untouched)
    #pragma unroll
    for (int mask = 1; mask < 16; mask <<= 1) {
        #pragma unroll
        for (int v = 0; v < 8; ++v)
            p[v] += __shfl_xor(p[v], mask, 32);
    }

    if (l16 == 0) {
        const float b2v  = b2[0];
        const int   base = mrow + half * 8;                // M = v + 8*half
        #pragma unroll
        for (int v = 0; v < 8; ++v)
            scores[base + v] = p[v] + b2v;
    }
}

// ---------------------------------------------------------------------------
// Kernel 2: per-(b,k) softmax stats -> c = exp(-m)/den  (0 for masked/empty).
// Then w[t] = exp(s[t]) * c exactly equals exp(s-m)/den (|s| <= ~8, safe).
// ---------------------------------------------------------------------------
__global__ void __launch_bounds__(256)
slot_stats_kernel(const float* __restrict__ scores,
                  const int*   __restrict__ bounds,
                  const int*   __restrict__ mask,
                  float*       __restrict__ cvec) {
    const int bk = blockIdx.x;           // b*K + k
    const int b  = bk >> 7;              // /K_
    const int s0 = bounds[bk * 2 + 0];
    const int s1 = bounds[bk * 2 + 1];
    const int tid = threadIdx.x;

    __shared__ float red[256];

    if (mask[bk] <= 0 || s0 >= s1) {
        if (tid == 0) cvec[bk] = 0.f;
        return;
    }

    const float* sp = scores + (size_t)b * T_;

    float lm = -INFINITY;
    for (int t = s0 + tid; t < s1; t += 256) lm = fmaxf(lm, sp[t]);
    red[tid] = lm;
    __syncthreads();
    for (int s = 128; s > 0; s >>= 1) {
        if (tid < s) red[tid] = fmaxf(red[tid], red[tid + s]);
        __syncthreads();
    }
    const float m = red[0];
    __syncthreads();

    float ls = 0.f;
    for (int t = s0 + tid; t < s1; t += 256) ls += __expf(sp[t] - m);
    red[tid] = ls;
    __syncthreads();
    for (int s = 128; s > 0; s >>= 1) {
        if (tid < s) red[tid] += red[tid + s];
        __syncthreads();
    }
    if (tid == 0) cvec[bk] = __expf(-m) / red[0];
}

// ---------------------------------------------------------------------------
// Kernel 3: out[b,k,h] = sum_{t in [s0,s1)} exp(scores[t])*c * X[b,t,h]
// Block per (b,k); thread h owns one output column; weights staged in LDS
// per 256-token chunk; next X chunk prefetched (global_prefetch_b8).
// ---------------------------------------------------------------------------
__global__ void __launch_bounds__(256)
pool_kernel(const float* __restrict__ X,
            const float* __restrict__ scores,
            const int*   __restrict__ bounds,
            const float* __restrict__ cvec,
            float*       __restrict__ out) {
    const int bk  = blockIdx.x;
    const int b   = bk >> 7;
    const int tid = threadIdx.x;         // h column
    const int s0  = bounds[bk * 2 + 0];
    const int s1  = bounds[bk * 2 + 1];
    const float c = cvec[bk];

    float* op = out + (size_t)bk * H_;
    if (c == 0.f) {                      // masked or empty slot -> zeros
        op[tid] = 0.f;
        return;
    }

    const float* sp = scores + (size_t)b * T_;
    __shared__ float wl[256];
    float acc = 0.f;

    for (int t0 = s0; t0 < s1; t0 += 256) {
        __syncthreads();
        const int t = t0 + tid;
        wl[tid] = (t < s1) ? __expf(sp[t]) * c : 0.f;
        __syncthreads();

        const int cnt = min(256, s1 - t0);
        const float* Xp = X + ((size_t)b * T_ + t0) * H_ + tid;

        if (t0 + 256 < s1)               // warm next chunk into cache
            __builtin_prefetch(X + ((size_t)b * T_ + t0 + 256) * H_ + tid, 0, 0);

        #pragma unroll 4
        for (int j = 0; j < cnt; ++j)
            acc = fmaf(wl[j], Xp[(size_t)j * H_], acc);
    }
    op[tid] = acc;
}

// ---------------------------------------------------------------------------
extern "C" void kernel_launch(void* const* d_in, const int* in_sizes, int n_in,
                              void* d_out, int out_size, void* d_ws, size_t ws_size,
                              hipStream_t stream) {
    const float* X   = (const float*)d_in[0];   // (B,T,H)
    const int*   bnd = (const int*)  d_in[1];   // (B,K,2)
    const int*   msk = (const int*)  d_in[2];   // (B,K)
    const float* W1  = (const float*)d_in[3];   // (H,HQ)
    const float* b1  = (const float*)d_in[4];   // (HQ,)
    const float* W2  = (const float*)d_in[5];   // (HQ,1)
    const float* b2  = (const float*)d_in[6];   // (1,)
    float* out = (float*)d_out;                 // (B,K,H)

    char* ws = (char*)d_ws;
    float*  scores = (float*)ws;                                   // B*T f32   (512 KB)
    __bf16* W1T    = (__bf16*)(ws + (size_t)B_ * T_ * 4);          // HQ*H bf16 ( 32 KB)
    float*  cvec   = (float*)(ws + (size_t)B_ * T_ * 4
                                 + (size_t)HQ_ * H_ * 2);          // B*K f32   (  8 KB)

    hipLaunchKernelGGL(w1_transpose_kernel, dim3(HQ_), dim3(H_), 0, stream, W1, W1T);
    hipLaunchKernelGGL(scores_wmma_kernel, dim3((B_ * T_) / 128), dim3(256), 0, stream,
                       X, W1T, b1, W2, b2, scores);
    hipLaunchKernelGGL(slot_stats_kernel, dim3(B_ * K_), dim3(256), 0, stream,
                       scores, bnd, msk, cvec);
    hipLaunchKernelGGL(pool_kernel, dim3(B_ * K_), dim3(256), 0, stream,
                       X, scores, bnd, cvec, out);
}
